// COGMENModel_32658931318920
// MI455X (gfx1250) — compile-verified, matrix operand order checked by reading, not run on hardware
//
#include <hip/hip_runtime.h>
#include <hip/hip_bf16.h>
#include <math.h>

// ---------------------------------------------------------------------------
// Model constants (fixed by the reference)
// ---------------------------------------------------------------------------
#define NTOK   4096
#define HDIM   256
#define NHEADS 4
#define NLAYERS 2
#define NREL   3
#define NCLS   6
#define NEDGES 131072

typedef __attribute__((ext_vector_type(16))) _Float16 v16h;
typedef __attribute__((ext_vector_type(8)))  _Float16 v8h;
typedef __attribute__((ext_vector_type(8)))  float    v8f;

// ---------------------------------------------------------------------------
// WMMA helpers
// ---------------------------------------------------------------------------
__device__ __forceinline__ v8f wmma_f16(v16h a, v16h b, v8f c) {
  return __builtin_amdgcn_wmma_f32_16x16x32_f16(
      /*neg_a=*/false, a, /*neg_b=*/false, b,
      /*c_mod=*/(short)0, c, /*reuse_a=*/false, /*reuse_b=*/false);
}

// Load a 16x32 f16 A/B fragment from an LDS tile stored K-contiguous with
// row stride `stride` halfs (stride must be a multiple of 8 for b128 loads).
// Layout (ISA 7.12.2): lane<16 -> row=lane,   K = {kk+0..7,  kk+16..23}
//                      lane>=16 -> row=lane-16, K = {kk+8..15, kk+24..31}
__device__ __forceinline__ v16h ldfrag(const _Float16* base, int row, int stride, int kk) {
  int lane = threadIdx.x & 31;
  const _Float16* p = base + (row + (lane & 15)) * stride + kk + ((lane >> 4) << 3);
  v8h lo = *(const v8h*)p;
  v8h hi = *(const v8h*)(p + 16);
  v16h r;
#pragma unroll
  for (int i = 0; i < 8; ++i) { r[i] = lo[i]; r[i + 8] = hi[i]; }
  return r;
}

// ---------------------------------------------------------------------------
// Generic WMMA GEMM: C[M,N] = act( (accum ? Cin : 0) + A[M,K] @ B + bias )
// B is [K,N] row-major (transB==0) or [N,K] row-major (transB==1).
// A/B are f32 in HBM, converted to f16 on the global->LDS path.
// 256 threads = 8 wave32, wave grid 4(M) x 2(N), each wave 2x2 16x16 tiles.
// ---------------------------------------------------------------------------
#define GBM 128
#define GBN 64
#define GBK 64
#define LSTR 72   // padded row stride (halfs); 144B -> conflict-free b128 frags

__global__ __launch_bounds__(256) void gemm_wmma_k(
    const float* __restrict__ A, const float* __restrict__ B,
    const float* __restrict__ bias, const float* __restrict__ Cin,
    float* __restrict__ C, int M, int N, int K, int transB, int accum, int act)
{
  __shared__ alignas(16) _Float16 As[GBM * LSTR];
  __shared__ alignas(16) _Float16 Bs[GBN * LSTR];

  const int t = threadIdx.x, lane = t & 31, w = t >> 5;
  const int wm = w >> 1, wn = w & 1;
  const int m0 = blockIdx.y * GBM, n0 = blockIdx.x * GBN;

  v8f acc[2][2] = {};

  for (int kk = 0; kk < K; kk += GBK) {
    // ---- A tile: 128x64 f32 -> f16 LDS (2048 float4, 8 per thread) ----
#pragma unroll
    for (int i = 0; i < 8; ++i) {
      int idx = t + i * 256;
      int r = idx >> 4, c = (idx & 15) << 2;
      int gm = m0 + r, gk = kk + c;
      float x0 = 0.f, x1 = 0.f, x2 = 0.f, x3 = 0.f;
      if (gm < M) {
        const float* p = A + (size_t)gm * K + gk;
        if (gk + 3 < K) {
          float4 v4 = *(const float4*)p;
          x0 = v4.x; x1 = v4.y; x2 = v4.z; x3 = v4.w;
        } else {
          if (gk     < K) x0 = p[0];
          if (gk + 1 < K) x1 = p[1];
          if (gk + 2 < K) x2 = p[2];
          if (gk + 3 < K) x3 = p[3];
        }
        if (i == 0 && gk + GBK < K) __builtin_prefetch(p + GBK, 0, 1);
      }
      _Float16* d = &As[r * LSTR + c];
      d[0] = (_Float16)x0; d[1] = (_Float16)x1; d[2] = (_Float16)x2; d[3] = (_Float16)x3;
    }
    // ---- B tile into Bs[n][k] (K-contiguous) ----
    if (transB) {             // B is [N,K]
#pragma unroll
      for (int i = 0; i < 4; ++i) {
        int idx = t + i * 256;
        int r = idx >> 4, c = (idx & 15) << 2;
        int gn = n0 + r, gk = kk + c;
        float x0 = 0.f, x1 = 0.f, x2 = 0.f, x3 = 0.f;
        if (gn < N) {
          const float* p = B + (size_t)gn * K + gk;
          if (gk + 3 < K) {
            float4 v4 = *(const float4*)p;
            x0 = v4.x; x1 = v4.y; x2 = v4.z; x3 = v4.w;
          } else {
            if (gk     < K) x0 = p[0];
            if (gk + 1 < K) x1 = p[1];
            if (gk + 2 < K) x2 = p[2];
            if (gk + 3 < K) x3 = p[3];
          }
        }
        _Float16* d = &Bs[r * LSTR + c];
        d[0] = (_Float16)x0; d[1] = (_Float16)x1; d[2] = (_Float16)x2; d[3] = (_Float16)x3;
      }
    } else {                  // B is [K,N]; transpose while staging
#pragma unroll
      for (int i = 0; i < 16; ++i) {
        int idx = t + i * 256;
        int kb = idx >> 6, nb = idx & 63;
        int gk = kk + kb, gn = n0 + nb;
        float v = (gk < K && gn < N) ? B[(size_t)gk * N + gn] : 0.f;
        Bs[nb * LSTR + kb] = (_Float16)v;
      }
    }
    __syncthreads();

#pragma unroll
    for (int kk2 = 0; kk2 < GBK; kk2 += 32) {
      v16h a0 = ldfrag(As, wm * 32,      LSTR, kk2);
      v16h a1 = ldfrag(As, wm * 32 + 16, LSTR, kk2);
      v16h b0 = ldfrag(Bs, wn * 32,      LSTR, kk2);
      v16h b1 = ldfrag(Bs, wn * 32 + 16, LSTR, kk2);
      acc[0][0] = wmma_f16(a0, b0, acc[0][0]);
      acc[0][1] = wmma_f16(a0, b1, acc[0][1]);
      acc[1][0] = wmma_f16(a1, b0, acc[1][0]);
      acc[1][1] = wmma_f16(a1, b1, acc[1][1]);
    }
    __syncthreads();
  }

  // ---- epilogue: C layout VGPR r -> M=r (lanes 0-15), M=r+8 (lanes 16-31) ----
#pragma unroll
  for (int mi = 0; mi < 2; ++mi) {
#pragma unroll
    for (int ni = 0; ni < 2; ++ni) {
      int gn = n0 + wn * 32 + ni * 16 + (lane & 15);
      int mb = m0 + wm * 32 + mi * 16 + ((lane >> 4) << 3);
#pragma unroll
      for (int r = 0; r < 8; ++r) {
        int gm = mb + r;
        if (gm < M && gn < N) {
          float v = acc[mi][ni][r];
          if (accum) v += Cin[(size_t)gm * N + gn];
          if (bias)  v += bias[gn];
          if (act == 1)      v = fmaxf(v, 0.f);
          else if (act == 2) v = 0.5f * v * (1.f + erff(v * 0.70710678118654752f));
          C[(size_t)gm * N + gn] = v;
        }
      }
    }
  }
}

// ---------------------------------------------------------------------------
// Flash attention, head dim 64, 64-query blocks, 64-key chunks, online softmax.
// grid = (NTOK/64, NHEADS), block = 256 (8 wave32, wave grid 4(M) x 2(N)).
// q is pre-scaled by 1/sqrt(64). qh/kh/vh/oh layout: [head][token][64].
// ---------------------------------------------------------------------------
#define FD   64
#define FSTR 72

__global__ __launch_bounds__(256) void flash_attn_k(
    const float* __restrict__ qh, const float* __restrict__ kh,
    const float* __restrict__ vh, float* __restrict__ oh, int Ntok)
{
  __shared__ alignas(16) _Float16 Qs[64 * FSTR];   // [row][d]
  __shared__ alignas(16) _Float16 Ks[64 * FSTR];   // [key][d]
  __shared__ alignas(16) _Float16 Vs[64 * FSTR];   // [d][key] (transposed)
  __shared__ alignas(16) _Float16 Ps[64 * FSTR];   // [row][key]
  __shared__ float Ss[64 * 66];                    // score staging
  __shared__ float row_max[64], row_sum[64], row_scale[64];

  const int t = threadIdx.x, lane = t & 31, w = t >> 5;
  const int wm = w >> 1, wn = w & 1;
  const int n0 = blockIdx.x * 64;
  const int h  = blockIdx.y;

  const float* qp    = qh + ((size_t)h * Ntok + n0) * FD;
  const float* kbase = kh + (size_t)h * Ntok * FD;
  const float* vbase = vh + (size_t)h * Ntok * FD;

  // Q tile (once)
#pragma unroll
  for (int i = 0; i < 4; ++i) {
    int idx = t + i * 256;
    int r = idx >> 4, c = (idx & 15) << 2;
    float4 v4 = *(const float4*)(qp + (size_t)r * FD + c);
    _Float16* d = &Qs[r * FSTR + c];
    d[0] = (_Float16)v4.x; d[1] = (_Float16)v4.y; d[2] = (_Float16)v4.z; d[3] = (_Float16)v4.w;
  }
  if (t < 64) { row_max[t] = -1e30f; row_sum[t] = 0.f; }
  v8f o0 = {}, o1 = {};
  __syncthreads();

  for (int kc = 0; kc < Ntok; kc += 64) {
    const float* kp = kbase + (size_t)kc * FD;
#pragma unroll
    for (int i = 0; i < 4; ++i) {
      int idx = t + i * 256;
      int r = idx >> 4, c = (idx & 15) << 2;
      float4 v4 = *(const float4*)(kp + (size_t)r * FD + c);
      _Float16* d = &Ks[r * FSTR + c];
      d[0] = (_Float16)v4.x; d[1] = (_Float16)v4.y; d[2] = (_Float16)v4.z; d[3] = (_Float16)v4.w;
    }
    const float* vp = vbase + (size_t)kc * FD;
#pragma unroll
    for (int i = 0; i < 16; ++i) {
      int idx = t + i * 256;
      int key = idx >> 6, dd = idx & 63;
      Vs[dd * FSTR + key] = (_Float16)vp[(size_t)key * FD + dd];
    }
    __syncthreads();

    // S = Q @ K^T  (per wave: 16 rows x 32 keys)
    v8f s0 = {}, s1 = {};
#pragma unroll
    for (int kk = 0; kk < 64; kk += 32) {
      v16h aq = ldfrag(Qs, wm * 16,      FSTR, kk);
      v16h b0 = ldfrag(Ks, wn * 32,      FSTR, kk);
      v16h b1 = ldfrag(Ks, wn * 32 + 16, FSTR, kk);
      s0 = wmma_f16(aq, b0, s0);
      s1 = wmma_f16(aq, b1, s1);
    }
    {
      int col = wn * 32 + (lane & 15);
      int rb  = wm * 16 + ((lane >> 4) << 3);
#pragma unroll
      for (int r = 0; r < 8; ++r) {
        Ss[(rb + r) * 66 + col]      = s0[r];
        Ss[(rb + r) * 66 + col + 16] = s1[r];
      }
    }
    __syncthreads();

    // online softmax update, one thread per row
    if (t < 64) {
      float om = row_max[t], cm = om;
#pragma unroll 8
      for (int c = 0; c < 64; ++c) cm = fmaxf(cm, Ss[t * 66 + c]);
      float corr = __expf(om - cm);
      float s = row_sum[t] * corr;
#pragma unroll 8
      for (int c = 0; c < 64; ++c) {
        float pv = __expf(Ss[t * 66 + c] - cm);
        s += pv;
        Ps[t * FSTR + c] = (_Float16)pv;
      }
      row_sum[t] = s; row_max[t] = cm; row_scale[t] = corr;
    }
    __syncthreads();

    // rescale O fragments, then O += P @ Vchunk
    {
      int rb = wm * 16 + ((lane >> 4) << 3);
#pragma unroll
      for (int r = 0; r < 8; ++r) {
        float sc = row_scale[rb + r];
        o0[r] *= sc; o1[r] *= sc;
      }
    }
#pragma unroll
    for (int kk = 0; kk < 64; kk += 32) {
      v16h ap = ldfrag(Ps, wm * 16,      FSTR, kk);
      v16h b0 = ldfrag(Vs, wn * 32,      FSTR, kk);
      v16h b1 = ldfrag(Vs, wn * 32 + 16, FSTR, kk);
      o0 = wmma_f16(ap, b0, o0);
      o1 = wmma_f16(ap, b1, o1);
    }
    __syncthreads();
  }

  // finalize O /= row_sum and store
  int col = wn * 32 + (lane & 15);
  int rb  = wm * 16 + ((lane >> 4) << 3);
#pragma unroll
  for (int r = 0; r < 8; ++r) {
    float inv = 1.f / row_sum[rb + r];
    float* op = oh + ((size_t)h * Ntok + n0 + rb + r) * FD;
    op[col]      = o0[r] * inv;
    op[col + 16] = o1[r] * inv;
  }
}

// ---------------------------------------------------------------------------
// Elementwise / graph kernels
// ---------------------------------------------------------------------------
__global__ void split_heads_k(const float* __restrict__ qkv,
                              float* __restrict__ qh, float* __restrict__ kh,
                              float* __restrict__ vh)
{
  int i = blockIdx.x * blockDim.x + threadIdx.x;
  if (i >= NTOK * 3 * HDIM) return;
  int n = i / (3 * HDIM), c = i % (3 * HDIM);
  int sec = c >> 8;          // 0=q 1=k 2=v
  int hc = c & 255;
  int h = hc >> 6, d = hc & 63;
  size_t dst = ((size_t)h * NTOK + n) * 64 + d;
  float v = qkv[i];
  if      (sec == 0) qh[dst] = v * 0.125f;   // fold 1/sqrt(64)
  else if (sec == 1) kh[dst] = v;
  else               vh[dst] = v;
}

__global__ void merge_heads_k(const float* __restrict__ oh, float* __restrict__ o)
{
  int i = blockIdx.x * blockDim.x + threadIdx.x;
  if (i >= NTOK * HDIM) return;
  int n = i >> 8, c = i & 255;
  int h = c >> 6, d = c & 63;
  o[i] = oh[((size_t)h * NTOK + n) * 64 + d];
}

__global__ __launch_bounds__(256) void layernorm_k(
    const float* __restrict__ in, const float* __restrict__ g,
    const float* __restrict__ b, float* __restrict__ out)
{
  __shared__ float red[256];
  int row = blockIdx.x, t = threadIdx.x;
  float v = in[(size_t)row * HDIM + t];
  red[t] = v;
  __syncthreads();
  for (int s = 128; s > 0; s >>= 1) { if (t < s) red[t] += red[t + s]; __syncthreads(); }
  float mu = red[0] * (1.f / HDIM);
  __syncthreads();
  float dv = v - mu;
  red[t] = dv * dv;
  __syncthreads();
  for (int s = 128; s > 0; s >>= 1) { if (t < s) red[t] += red[t + s]; __syncthreads(); }
  float var = red[0] * (1.f / HDIM);
  out[(size_t)row * HDIM + t] = dv * rsqrtf(var + 1e-5f) * g[t] + b[t];
}

// RGCN: per-edge message scatter. One wave per edge (8 elems/lane).
__global__ __launch_bounds__(256) void rgcn_scatter_k(
    const int* __restrict__ edge_index, const int* __restrict__ edge_type,
    const float* __restrict__ xr, float* __restrict__ sums, float* __restrict__ cnts)
{
  int w = threadIdx.x >> 5, lane = threadIdx.x & 31;
  int e = blockIdx.x * 8 + w;
  if (e >= NEDGES) return;
  int s  = edge_index[e];
  int dd = edge_index[NEDGES + e];
  int ty = edge_type[e];
  const float* msg = xr + ((size_t)ty * NTOK + s) * HDIM;
  float* dst = sums + ((size_t)dd * NREL + ty) * HDIM;
#pragma unroll
  for (int j = 0; j < 8; ++j) {
    int d = lane + j * 32;
    atomicAdd(&dst[d], msg[d]);
  }
  if (lane == 0) atomicAdd(&cnts[(size_t)dd * NREL + ty], 1.f);
}

__global__ void rgcn_combine_k(const float* __restrict__ sums,
                               const float* __restrict__ cnts,
                               const float* __restrict__ rootout,
                               const float* __restrict__ bias,
                               float* __restrict__ g)
{
  int i = blockIdx.x * blockDim.x + threadIdx.x;
  if (i >= NTOK * HDIM) return;
  int n = i >> 8, d = i & 255;
  float acc = rootout[i] + bias[d];
#pragma unroll
  for (int r = 0; r < NREL; ++r) {
    float c = cnts[(size_t)n * NREL + r];
    acc += sums[((size_t)n * NREL + r) * HDIM + d] / fmaxf(c, 1.f);
  }
  g[i] = fmaxf(acc, 0.f);
}

// order-preserving float<->uint map for atomicMax on floats
__device__ __forceinline__ unsigned int f2o(float f) {
  unsigned int b = __float_as_uint(f);
  return (b & 0x80000000u) ? ~b : (b | 0x80000000u);
}
__device__ __forceinline__ float o2f(unsigned int u) {
  unsigned int b = (u & 0x80000000u) ? (u & 0x7FFFFFFFu) : ~u;
  return __uint_as_float(b);
}

// edge attention logits: one wave per (edge, head); dot over 256 dims
__global__ __launch_bounds__(256) void edge_alpha_k(
    const int* __restrict__ edge_index, const float* __restrict__ q,
    const float* __restrict__ k, float* __restrict__ alpha,
    unsigned int* __restrict__ mu)
{
  int w = threadIdx.x >> 5, lane = threadIdx.x & 31;
  int p = blockIdx.x * 8 + w;
  if (p >= NEDGES * NHEADS) return;
  int e = p >> 2, h = p & 3;
  int s  = edge_index[e];
  int dd = edge_index[NEDGES + e];
  const float* qd = q + (size_t)dd * (NHEADS * HDIM) + h * HDIM;
  const float* ks = k + (size_t)s  * (NHEADS * HDIM) + h * HDIM;
  float acc = 0.f;
#pragma unroll
  for (int j = 0; j < 8; ++j) { int d = lane + j * 32; acc += qd[d] * ks[d]; }
  for (int off = 16; off > 0; off >>= 1) acc += __shfl_xor(acc, off, 32);
  if (lane == 0) {
    float a = acc * 0.0625f;   // 1/sqrt(256)
    alpha[p] = a;
    atomicMax(&mu[(size_t)dd * NHEADS + h], f2o(a));
  }
}

__global__ void edge_exp_k(const int* __restrict__ edge_index,
                           const float* __restrict__ alpha,
                           const unsigned int* __restrict__ mu,
                           float* __restrict__ earr, float* __restrict__ den)
{
  int i = blockIdx.x * blockDim.x + threadIdx.x;
  if (i >= NEDGES * NHEADS) return;
  int e = i >> 2, h = i & 3;
  int dd = edge_index[NEDGES + e];
  float m = o2f(mu[(size_t)dd * NHEADS + h]);
  float ee = __expf(alpha[i] - m);
  earr[i] = ee;
  atomicAdd(&den[(size_t)dd * NHEADS + h], ee);
}

// weighted aggregation: one wave per (edge, head)
__global__ __launch_bounds__(256) void edge_agg_k(
    const int* __restrict__ edge_index, const float* __restrict__ earr,
    const float* __restrict__ den, const float* __restrict__ v,
    float* __restrict__ agg)
{
  int w = threadIdx.x >> 5, lane = threadIdx.x & 31;
  int e = blockIdx.x * 2 + (w >> 2);
  int h = w & 3;
  if (e >= NEDGES) return;
  int s  = edge_index[e];
  int dd = edge_index[NEDGES + e];
  float wt = earr[(size_t)e * NHEADS + h] / den[(size_t)dd * NHEADS + h];
  const float* vp = v + (size_t)s * (NHEADS * HDIM) + h * HDIM;
  float* ap = agg + (size_t)dd * (NHEADS * HDIM) + h * HDIM;
#pragma unroll
  for (int j = 0; j < 8; ++j) {
    int d = lane + j * 32;
    atomicAdd(&ap[d], wt * vp[d]);
  }
}

__global__ void gt_combine_k(const float* __restrict__ agg,
                             const float* __restrict__ skipout,
                             float* __restrict__ g2)
{
  int i = blockIdx.x * blockDim.x + threadIdx.x;
  if (i >= NTOK * HDIM) return;
  int n = i >> 8, d = i & 255;
  const float* a = agg + (size_t)n * (NHEADS * HDIM);
  g2[i] = 0.25f * (a[d] + a[HDIM + d] + a[2 * HDIM + d] + a[3 * HDIM + d]) + skipout[i];
}

// ---------------------------------------------------------------------------
// Host-side orchestration
// ---------------------------------------------------------------------------
static inline void gemm(hipStream_t s, const float* A, const float* B,
                        const float* bias, const float* Cin, float* C,
                        int M, int N, int K, int accum, int act)
{
  dim3 g((N + GBN - 1) / GBN, (M + GBM - 1) / GBM);
  gemm_wmma_k<<<g, 256, 0, s>>>(A, B, bias, Cin, C, M, N, K, /*transB=*/0, accum, act);
}

extern "C" void kernel_launch(void* const* d_in, const int* in_sizes, int n_in,
                              void* d_out, int out_size, void* d_ws, size_t ws_size,
                              hipStream_t stream)
{
  const float* text  = (const float*)d_in[0];
  const float* audio = (const float*)d_in[1];
  const float* vis   = (const float*)d_in[2];
  const int* edge_index = (const int*)d_in[3];
  const int* edge_type  = (const int*)d_in[4];
  const float* w_text = (const float*)d_in[5];
  const float* b_text = (const float*)d_in[6];
  const float* w_audio = (const float*)d_in[7];
  const float* b_audio = (const float*)d_in[8];
  const float* w_vis = (const float*)d_in[9];
  const float* b_vis = (const float*)d_in[10];
  const float* enc_wqkv = (const float*)d_in[11];
  const float* enc_bqkv = (const float*)d_in[12];
  const float* enc_wo = (const float*)d_in[13];
  const float* enc_bo = (const float*)d_in[14];
  const float* enc_ln1_g = (const float*)d_in[15];
  const float* enc_ln1_b = (const float*)d_in[16];
  const float* enc_w1 = (const float*)d_in[17];
  const float* enc_b1 = (const float*)d_in[18];
  const float* enc_w2 = (const float*)d_in[19];
  const float* enc_b2 = (const float*)d_in[20];
  const float* enc_ln2_g = (const float*)d_in[21];
  const float* enc_ln2_b = (const float*)d_in[22];
  const float* rgcn_root = (const float*)d_in[23];
  const float* rgcn_rel  = (const float*)d_in[24];
  const float* rgcn_bias = (const float*)d_in[25];
  const float* gt_wq = (const float*)d_in[26];
  const float* gt_bq = (const float*)d_in[27];
  const float* gt_wk = (const float*)d_in[28];
  const float* gt_bk = (const float*)d_in[29];
  const float* gt_wv = (const float*)d_in[30];
  const float* gt_bv = (const float*)d_in[31];
  const float* gt_wskip = (const float*)d_in[32];
  const float* gt_bskip = (const float*)d_in[33];
  const float* cls_w1 = (const float*)d_in[34];
  const float* cls_b1 = (const float*)d_in[35];
  const float* cls_w2 = (const float*)d_in[36];
  const float* cls_b2 = (const float*)d_in[37];
  float* out = (float*)d_out;

  // workspace bump allocator (256B aligned)
  char* wp = (char*)d_ws;
  auto alloc = [&](size_t nfloats) -> float* {
    float* r = (float*)wp;
    wp += ((nfloats * sizeof(float) + 255) & ~(size_t)255);
    return r;
  };
  float* x    = alloc((size_t)NTOK * HDIM);
  float* tmp  = alloc((size_t)NTOK * HDIM);
  float* qkv  = alloc((size_t)NTOK * 3 * HDIM);
  float* qh   = alloc((size_t)NHEADS * NTOK * 64);
  float* kh   = alloc((size_t)NHEADS * NTOK * 64);
  float* vh   = alloc((size_t)NHEADS * NTOK * 64);
  float* oh   = alloc((size_t)NHEADS * NTOK * 64);
  float* omrg = alloc((size_t)NTOK * HDIM);
  float* ff   = alloc((size_t)NTOK * 4 * HDIM);
  float* xr   = alloc((size_t)NREL * NTOK * HDIM);
  float* sums = alloc((size_t)NTOK * NREL * HDIM);
  float* cnts = alloc((size_t)NTOK * NREL);
  float* rootout = alloc((size_t)NTOK * HDIM);
  float* gbuf = alloc((size_t)NTOK * HDIM);
  float* gtq  = alloc((size_t)NTOK * NHEADS * HDIM);
  float* gtk  = alloc((size_t)NTOK * NHEADS * HDIM);
  float* gtv  = alloc((size_t)NTOK * NHEADS * HDIM);
  float* skipout = alloc((size_t)NTOK * HDIM);
  float* alphab  = alloc((size_t)NEDGES * NHEADS);
  float* earr = alloc((size_t)NEDGES * NHEADS);
  unsigned int* mu = (unsigned int*)alloc((size_t)NTOK * NHEADS);
  float* den  = alloc((size_t)NTOK * NHEADS);
  float* agg  = alloc((size_t)NTOK * NHEADS * HDIM);
  float* g2   = alloc((size_t)NTOK * HDIM);
  float* clsh = alloc((size_t)NTOK * HDIM);
  (void)ws_size; (void)in_sizes; (void)n_in; (void)out_size;

  // --- fused modality projections: x = T@Wt + A@Wa + V@Wv + biases ---
  gemm(stream, text,  w_text,  b_text,  nullptr, x, NTOK, HDIM, 768, 0, 0);
  gemm(stream, audio, w_audio, b_audio, x,       x, NTOK, HDIM, 100, 1, 0);
  gemm(stream, vis,   w_vis,   b_vis,   x,       x, NTOK, HDIM, 512, 1, 0);

  // --- transformer encoder ---
  for (int l = 0; l < NLAYERS; ++l) {
    const float* wqkv = enc_wqkv + (size_t)l * HDIM * 3 * HDIM;
    const float* bqkv = enc_bqkv + (size_t)l * 3 * HDIM;
    const float* wo = enc_wo + (size_t)l * HDIM * HDIM;
    const float* bo = enc_bo + (size_t)l * HDIM;
    const float* w1 = enc_w1 + (size_t)l * HDIM * 4 * HDIM;
    const float* b1 = enc_b1 + (size_t)l * 4 * HDIM;
    const float* w2 = enc_w2 + (size_t)l * 4 * HDIM * HDIM;
    const float* b2 = enc_b2 + (size_t)l * HDIM;

    gemm(stream, x, wqkv, bqkv, nullptr, qkv, NTOK, 3 * HDIM, HDIM, 0, 0);
    split_heads_k<<<(NTOK * 3 * HDIM + 255) / 256, 256, 0, stream>>>(qkv, qh, kh, vh);
    flash_attn_k<<<dim3(NTOK / 64, NHEADS), 256, 0, stream>>>(qh, kh, vh, oh, NTOK);
    merge_heads_k<<<(NTOK * HDIM + 255) / 256, 256, 0, stream>>>(oh, omrg);
    gemm(stream, omrg, wo, bo, x, tmp, NTOK, HDIM, HDIM, 1, 0);    // x + attn
    layernorm_k<<<NTOK, 256, 0, stream>>>(tmp, enc_ln1_g + (size_t)l * HDIM,
                                          enc_ln1_b + (size_t)l * HDIM, x);
    gemm(stream, x, w1, b1, nullptr, ff, NTOK, 4 * HDIM, HDIM, 0, 2);  // gelu
    gemm(stream, ff, w2, b2, x, tmp, NTOK, HDIM, 4 * HDIM, 1, 0);      // x + ffn
    layernorm_k<<<NTOK, 256, 0, stream>>>(tmp, enc_ln2_g + (size_t)l * HDIM,
                                          enc_ln2_b + (size_t)l * HDIM, x);
  }

  // --- RGCN ---
  for (int r = 0; r < NREL; ++r)
    gemm(stream, x, rgcn_rel + (size_t)r * HDIM * HDIM, nullptr, nullptr,
         xr + (size_t)r * NTOK * HDIM, NTOK, HDIM, HDIM, 0, 0);
  gemm(stream, x, rgcn_root, nullptr, nullptr, rootout, NTOK, HDIM, HDIM, 0, 0);
  hipMemsetAsync(sums, 0, (size_t)NTOK * NREL * HDIM * sizeof(float), stream);
  hipMemsetAsync(cnts, 0, (size_t)NTOK * NREL * sizeof(float), stream);
  rgcn_scatter_k<<<(NEDGES + 7) / 8, 256, 0, stream>>>(edge_index, edge_type, xr, sums, cnts);
  rgcn_combine_k<<<(NTOK * HDIM + 255) / 256, 256, 0, stream>>>(sums, cnts, rootout,
                                                                rgcn_bias, gbuf);

  // --- graph TransformerConv ---
  gemm(stream, gbuf, gt_wq, gt_bq, nullptr, gtq, NTOK, NHEADS * HDIM, HDIM, 0, 0);
  gemm(stream, gbuf, gt_wk, gt_bk, nullptr, gtk, NTOK, NHEADS * HDIM, HDIM, 0, 0);
  gemm(stream, gbuf, gt_wv, gt_bv, nullptr, gtv, NTOK, NHEADS * HDIM, HDIM, 0, 0);
  gemm(stream, gbuf, gt_wskip, gt_bskip, nullptr, skipout, NTOK, HDIM, HDIM, 0, 0);
  hipMemsetAsync(mu, 0, (size_t)NTOK * NHEADS * sizeof(unsigned int), stream);
  hipMemsetAsync(den, 0, (size_t)NTOK * NHEADS * sizeof(float), stream);
  hipMemsetAsync(agg, 0, (size_t)NTOK * NHEADS * HDIM * sizeof(float), stream);
  edge_alpha_k<<<(NEDGES * NHEADS + 7) / 8, 256, 0, stream>>>(edge_index, gtq, gtk, alphab, mu);
  edge_exp_k<<<(NEDGES * NHEADS + 255) / 256, 256, 0, stream>>>(edge_index, alphab, mu, earr, den);
  edge_agg_k<<<(NEDGES + 1) / 2, 256, 0, stream>>>(edge_index, earr, den, gtv, agg);
  gt_combine_k<<<(NTOK * HDIM + 255) / 256, 256, 0, stream>>>(agg, skipout, g2);

  // --- classifier ---
  gemm(stream, g2, cls_w1, cls_b1, nullptr, clsh, NTOK, HDIM, HDIM, 0, 1);   // relu
  gemm(stream, clsh, cls_w2, cls_b2, nullptr, out, NTOK, NCLS, HDIM, 0, 0);
}
